// LNO2d_43671227465837
// MI455X (gfx1250) — matrix operand order repaired and checked by reading, not compiled, not run on hardware
//
#include <hip/hip_runtime.h>

// ---------------------------------------------------------------------------
// LNO2d for MI455X (gfx1250, wave32). GEMM-shaped work runs on
// v_wmma_f32_16x16x32_bf16 (fp32 accumulate); glue ops are VALU/LDS.
// ---------------------------------------------------------------------------

typedef __bf16 bf16_t;
typedef bf16_t v16bf __attribute__((ext_vector_type(16)));
typedef float  v8f   __attribute__((ext_vector_type(8)));

union FragU { v16bf v; uint4 q[2]; };

__device__ __forceinline__ unsigned short f2bf(float f) {
    union { float f; unsigned u; } v; v.f = f;
    unsigned u = v.u;
    unsigned r = u + 0x7FFFu + ((u >> 16) & 1u);   // round-to-nearest-even
    return (unsigned short)(r >> 16);
}

// A fragment: 16x32 bf16 tile at src (row-major, ld elements/row).
// lane<16: K = [0..7, 16..23]; lane>=16: K = [8..15, 24..31]; M = lane&15.
__device__ __forceinline__ v16bf load_frag_a(const unsigned short* src, int ld, int lane) {
    int m = lane & 15;
    int o = (lane < 16) ? 0 : 8;
    const unsigned short* p = src + (size_t)m * ld;
    FragU f;
    f.q[0] = *(const uint4*)(p + o);
    f.q[1] = *(const uint4*)(p + 16 + o);
    return f.v;
}

// B fragment: 32x16 bf16 tile, weights packed [N][K] (K contiguous, ldk).
// lane&15 = output column; lane<16 holds K=[0..7,16..23], lane>=16 holds K=[8..15,24..31].
__device__ __forceinline__ v16bf load_frag_b(const unsigned short* w, int ldk, int lane) {
    int nn = lane & 15;
    int ko = (lane < 16) ? 0 : 8;
    const unsigned short* p = w + (size_t)nn * ldk + ko;
    FragU f;
    f.q[0] = *(const uint4*)(p);
    f.q[1] = *(const uint4*)(p + 16);
    return f.v;
}

__device__ __forceinline__ v8f wmma_bf16(v16bf a, v16bf b, v8f c) {
    return __builtin_amdgcn_wmma_f32_16x16x32_bf16(false, a, false, b, (short)0, c, false, false);
}

// ---------------------------------------------------------------------------
// Weight prepack kernels (fp32 -> bf16, GEMM-friendly layouts)
// ---------------------------------------------------------------------------

// convW (NB=4, L=3, O=64, I=64, 3, 3) -> [i][l][co][tap][ci] bf16
__global__ void k_pack_conv(const float* __restrict__ src, unsigned short* __restrict__ dst) {
    int tid = blockIdx.x * blockDim.x + threadIdx.x;
    const int total = 4 * 3 * 64 * 64 * 9;
    if (tid >= total) return;
    int t = tid;
    int kx = t % 3; t /= 3;
    int ky = t % 3; t /= 3;
    int ci = t % 64; t /= 64;
    int co = t % 64; t /= 64;
    int l  = t % 3;  t /= 3;
    int i  = t;
    size_t s = ((((size_t)(i * 3 + l) * 64 + co) * 64 + ci) * 3 + ky) * 3 + kx;
    size_t d = (((size_t)(i * 3 + l) * 64 + co) * 9 + (ky * 3 + kx)) * 64 + ci;
    dst[d] = f2bf(src[s]);
}

// fp32 [K][N] -> bf16 [N][K]
__global__ void k_pack_mat(const float* __restrict__ src, unsigned short* __restrict__ dst,
                           int K, int N) {
    int tid = blockIdx.x * blockDim.x + threadIdx.x;
    if (tid >= K * N) return;
    int k = tid / N, n = tid % N;
    dst[(size_t)n * K + k] = f2bf(src[(size_t)k * N + n]);
}

// ---------------------------------------------------------------------------
// Glue kernels
// ---------------------------------------------------------------------------

// h = [a0,a1,x] @ pW + pb     (one block per pixel, 64 threads = channels)
__global__ void k_lift(const float* __restrict__ a, const float* __restrict__ x,
                       const float* __restrict__ pW, const float* __restrict__ pb,
                       float* __restrict__ h) {
    size_t p = blockIdx.x;
    int c = threadIdx.x;
    float a0 = a[p * 2 + 0], a1 = a[p * 2 + 1], xv = x[p];
    h[p * 64 + c] = a0 * pW[c] + a1 * pW[64 + c] + xv * pW[128 + c] + pb[c];
}

// bf16 (optionally strided-downsampled) copy of h. Grid = B*Sout*Sout blocks.
__global__ void k_cvt_ds(const float* __restrict__ src, unsigned short* __restrict__ dst,
                         int Sout, int st) {
    size_t p = blockIdx.x;
    int c = threadIdx.x;
    int per = Sout * Sout;
    int b = (int)(p / per), rem = (int)(p % per);
    int yi = rem / Sout, xi = rem % Sout;
    size_t s = (((size_t)b * 256 + (size_t)yi * st) * 256 + (size_t)xi * st) * 64 + c;
    dst[p * 64 + c] = f2bf(src[s]);
}

__device__ __forceinline__ float bilin64(const float* __restrict__ src, int Hl, int Wl,
                                         int b, float sy, float sx, int c) {
    int y0 = (int)floorf(sy), x0 = (int)floorf(sx);
    float fy = sy - (float)y0, fx = sx - (float)x0;
    int y1 = y0 + 1, x1 = x0 + 1;
    y0 = min(max(y0, 0), Hl - 1); y1 = min(max(y1, 0), Hl - 1);
    x0 = min(max(x0, 0), Wl - 1); x1 = min(max(x1, 0), Wl - 1);
    const float* base = src + (size_t)b * Hl * Wl * 64;
    float v00 = base[((size_t)y0 * Wl + x0) * 64 + c];
    float v01 = base[((size_t)y0 * Wl + x1) * 64 + c];
    float v10 = base[((size_t)y1 * Wl + x0) * 64 + c];
    float v11 = base[((size_t)y1 * Wl + x1) * 64 + c];
    return v00 * (1.f - fy) * (1.f - fx) + v01 * (1.f - fy) * fx
         + v10 * fy * (1.f - fx) + v11 * fy * fx;
}

// hi[b,y,x,c] += bilinear(lo)   (hi is 256x256, lo is Hl x Hl)
__global__ void k_upsample_add(const float* __restrict__ lo, float* __restrict__ hi, int Hl) {
    size_t p = blockIdx.x;
    int c = threadIdx.x;
    int b = (int)(p >> 16), rem = (int)(p & 65535);
    int y = rem >> 8, xx = rem & 255;
    float sc = (float)Hl / 256.0f;
    float sy = ((float)y + 0.5f) * sc - 0.5f;
    float sx = ((float)xx + 0.5f) * sc - 0.5f;
    hi[p * 64 + c] += bilin64(lo, Hl, Hl, b, sy, sx, c);
}

// phi/psi layer 1: 3 -> 64, relu, bf16 out. Grid = 32768 pts, 64 thr.
__global__ void k_mlp1(const float* __restrict__ a, const float* __restrict__ x,
                       const float* __restrict__ W1, const float* __restrict__ b1,
                       unsigned short* __restrict__ out) {
    size_t p = blockIdx.x;
    int j = threadIdx.x;
    int b = (int)(p >> 14), rem = (int)(p & 16383);
    int yi = rem >> 7, xi = rem & 127;
    size_t s = ((size_t)b * 256 + yi * 2) * 256 + xi * 2;
    float f0 = a[s * 2 + 0], f1 = a[s * 2 + 1], f2 = x[s];
    float hv = f0 * W1[j] + f1 * W1[64 + j] + f2 * W1[128 + j] + b1[j];
    out[p * 64 + j] = f2bf(fmaxf(hv, 0.f));
}

// y[b,w,r] = (1/n) sum_p psi[b,p,w,r] * h[b,2yi,2xi,w]     grid=(64,B), 256 thr
__global__ void k_reduce_y(const float* __restrict__ psi, const float* __restrict__ h,
                           float* __restrict__ y) {
    int w = blockIdx.x, b = blockIdx.y, t = threadIdx.x;
    float a0 = 0.f, a1 = 0.f, a2 = 0.f, a3 = 0.f;
    for (int p = t; p < 16384; p += 256) {
        int yi = p >> 7, xi = p & 127;
        float v = h[(((size_t)b * 256 + (yi << 1)) * 256 + (xi << 1)) * 64 + w];
        const float* pp = psi + ((size_t)(b * 16384 + p)) * 256 + (w << 2);
        a0 += pp[0] * v; a1 += pp[1] * v; a2 += pp[2] * v; a3 += pp[3] * v;
    }
    __shared__ float s0[256], s1[256], s2[256], s3[256];
    s0[t] = a0; s1[t] = a1; s2[t] = a2; s3[t] = a3;
    __syncthreads();
    for (int st = 128; st > 0; st >>= 1) {
        if (t < st) { s0[t] += s0[t + st]; s1[t] += s1[t + st];
                      s2[t] += s2[t + st]; s3[t] += s3[t + st]; }
        __syncthreads();
    }
    if (t == 0) {
        const float inv = 1.0f / 16384.0f;
        float* yy = y + ((size_t)(b * 64 + w) << 2);
        yy[0] = s0[0] * inv; yy[1] = s1[0] * inv; yy[2] = s2[0] * inv; yy[3] = s3[0] * inv;
    }
}

// x2lo[b,p,w] = sum_r phi[b,p,w,r]*y[b,w,r]     grid = 32768 pts, 64 thr
__global__ void k_x2(const float* __restrict__ phi, const float* __restrict__ y,
                     float* __restrict__ x2lo) {
    size_t p = blockIdx.x;
    int c = threadIdx.x;
    int b = (int)(p >> 14);
    const float* ph = phi + p * 256 + (c << 2);
    const float* yy = y + ((size_t)(b * 64 + c) << 2);
    x2lo[p * 64 + c] = ph[0] * yy[0] + ph[1] * yy[1] + ph[2] * yy[2] + ph[3] * yy[3];
}

// h = LN(x1 + up2(x2lo)) * g + b, optional relu.  One block/pixel, 64 thr.
__global__ void k_combine_ln(const float* __restrict__ x1, const float* __restrict__ x2lo,
                             const float* __restrict__ g, const float* __restrict__ bb,
                             float* __restrict__ h, int relu) {
    size_t p = blockIdx.x;
    int c = threadIdx.x;
    int b = (int)(p >> 16), rem = (int)(p & 65535);
    int y = rem >> 8, xx = rem & 255;
    float sy = ((float)y + 0.5f) * 0.5f - 0.5f;
    float sx = ((float)xx + 0.5f) * 0.5f - 0.5f;
    float val = x1[p * 64 + c] + bilin64(x2lo, 128, 128, b, sy, sx, c);
    __shared__ float sv[64], sq[64];
    sv[c] = val; sq[c] = val * val;
    __syncthreads();
    for (int st = 32; st > 0; st >>= 1) {
        if (c < st) { sv[c] += sv[c + st]; sq[c] += sq[c + st]; }
        __syncthreads();
    }
    float mu  = sv[0] * (1.0f / 64.0f);
    float var = sq[0] * (1.0f / 64.0f) - mu * mu;
    float o = (val - mu) * rsqrtf(var + 1e-5f) * g[c] + bb[c];
    h[p * 64 + c] = relu ? fmaxf(o, 0.f) : o;
}

// out[p] = qh1[p,:] . q2W + q2b      one thread per pixel
__global__ void k_q2(const float* __restrict__ qh1, const float* __restrict__ q2W,
                     const float* __restrict__ q2b, float* __restrict__ out, int Np) {
    int p = blockIdx.x * blockDim.x + threadIdx.x;
    if (p >= Np) return;
    const float* hp = qh1 + (size_t)p * 128;
    float acc = q2b[0];
    for (int j = 0; j < 128; ++j) acc += hp[j] * q2W[j];
    out[p] = acc;
}

// ---------------------------------------------------------------------------
// WMMA kernels
// ---------------------------------------------------------------------------

// 3x3 SAME conv, 64->64 ch, NHWC bf16 in, fp32 out. Block = 128 thr = 4 waves,
// each wave: 16 pixels x 16 out-channels; 9 taps x 2 K-chunks = 18 WMMAs.
// grid = (W/16, H, B). A tile staged in LDS (3 rows x 18 px x 64 ci).
__global__ __launch_bounds__(128)
void k_conv3x3_wmma(const unsigned short* __restrict__ X, const unsigned short* __restrict__ Wp,
                    const float* __restrict__ bias, float* __restrict__ Out, int H) {
    const int W = H;
    __shared__ unsigned short sX[3 * 18 * 64];
    int b = blockIdx.z, y = blockIdx.y, x0 = blockIdx.x * 16;

    for (int e = threadIdx.x; e < 3 * 18 * 64; e += 128) {
        int r  = e / (18 * 64);
        int rm = e % (18 * 64);
        int px = rm / 64, ci = rm % 64;
        int yy = y + r - 1, xx = x0 + px - 1;
        unsigned short v = 0;
        if (yy >= 0 && yy < H && xx >= 0 && xx < W)
            v = X[(((size_t)b * H + yy) * W + xx) * 64 + ci];
        sX[e] = v;
    }
    __syncthreads();

    int wave = threadIdx.x >> 5, lane = threadIdx.x & 31;
    int co0 = wave * 16;
    const unsigned short* wb = Wp + (size_t)co0 * 576;   // [co][tap][ci]

    v8f acc = {};
    #pragma unroll
    for (int t = 0; t < 9; ++t) {
        int dy = t / 3, dx = t % 3;                      // 0..2 -> rows already offset
        #pragma unroll
        for (int kc = 0; kc < 2; ++kc) {
            v16bf fa = load_frag_a(&sX[(dy * 18 + dx) * 64 + kc * 32], 64, lane);
            v16bf fb = load_frag_b(wb + t * 64 + kc * 32, 576, lane);
            acc = wmma_bf16(fa, fb, acc);
        }
    }

    int n = lane & 15;
    int mofs = (lane < 16) ? 0 : 8;
    float bv = bias[co0 + n];
    #pragma unroll
    for (int v = 0; v < 8; ++v) {
        int px = x0 + v + mofs;
        Out[(((size_t)b * H + y) * W + px) * 64 + co0 + n] = acc[v] + bv;
    }
}

// C[M,N] = relu?(A[M,K] @ Wp[N,K]^T + bias). A,Wp bf16; Cf fp32 / Cb bf16 optional.
// Block = 128 thr = 4 waves; tile = 16 rows x 64 cols. grid = (M/16, N/64).
__global__ __launch_bounds__(128)
void k_gemm_wmma(const unsigned short* __restrict__ A, const unsigned short* __restrict__ Wp,
                 const float* __restrict__ bias, float* __restrict__ Cf,
                 unsigned short* __restrict__ Cb, int N, int K, int relu) {
    int wave = threadIdx.x >> 5, lane = threadIdx.x & 31;
    size_t row0 = (size_t)blockIdx.x * 16;
    int n0 = blockIdx.y * 64 + wave * 16;

    v8f acc = {};
    for (int kc = 0; kc < K; kc += 32) {
        v16bf fa = load_frag_a(A + row0 * K + kc, K, lane);
        v16bf fb = load_frag_b(Wp + (size_t)n0 * K + kc, K, lane);
        acc = wmma_bf16(fa, fb, acc);
    }

    int n = lane & 15;
    int mofs = (lane < 16) ? 0 : 8;
    float bv = bias[n0 + n];
    #pragma unroll
    for (int v = 0; v < 8; ++v) {
        size_t r = row0 + v + mofs;
        float val = acc[v] + bv;
        if (relu) val = fmaxf(val, 0.f);
        if (Cf) Cf[r * N + n0 + n] = val;
        if (Cb) Cb[r * N + n0 + n] = f2bf(val);
    }
}

// ---------------------------------------------------------------------------
// Host orchestration
// ---------------------------------------------------------------------------

static inline size_t alignup(size_t x) { return (x + 255) & ~(size_t)255; }

extern "C" void kernel_launch(void* const* d_in, const int* in_sizes, int n_in,
                              void* d_out, int out_size, void* d_ws, size_t ws_size,
                              hipStream_t stream) {
    (void)in_sizes; (void)n_in; (void)out_size; (void)ws_size;

    const float* x     = (const float*)d_in[0];
    const float* a     = (const float*)d_in[1];
    const float* pW    = (const float*)d_in[2];
    const float* pb    = (const float*)d_in[3];
    const float* q1W   = (const float*)d_in[4];
    const float* q1b   = (const float*)d_in[5];
    const float* q2W   = (const float*)d_in[6];
    const float* q2b   = (const float*)d_in[7];
    const float* phiW1 = (const float*)d_in[8];
    const float* phib1 = (const float*)d_in[9];
    const float* phiW2 = (const float*)d_in[10];
    const float* phib2 = (const float*)d_in[11];
    const float* phiW3 = (const float*)d_in[12];
    const float* phib3 = (const float*)d_in[13];
    const float* psiW1 = (const float*)d_in[14];
    const float* psib1 = (const float*)d_in[15];
    const float* psiW2 = (const float*)d_in[16];
    const float* psib2 = (const float*)d_in[17];
    const float* psiW3 = (const float*)d_in[18];
    const float* psib3 = (const float*)d_in[19];
    const float* lnG   = (const float*)d_in[20];
    const float* lnB   = (const float*)d_in[21];
    const float* convW = (const float*)d_in[22];
    const float* convb = (const float*)d_in[23];
    float* out = (float*)d_out;

    char* ws = (char*)d_ws;
    size_t off = 0;
    auto take = [&](size_t bytes) { size_t o = off; off += alignup(bytes); return o; };

    size_t o_convW = take((size_t)4 * 3 * 64 * 9 * 64 * 2);
    size_t o_phi2W = take((size_t)4 * 128 * 64 * 2);
    size_t o_phi3W = take((size_t)4 * 256 * 128 * 2);
    size_t o_psi2W = take((size_t)4 * 128 * 64 * 2);
    size_t o_psi3W = take((size_t)4 * 256 * 128 * 2);
    size_t o_q1W   = take((size_t)128 * 64 * 2);
    size_t o_h     = take((size_t)2 * 256 * 256 * 64 * 4);
    size_t o_x1    = take((size_t)2 * 256 * 256 * 64 * 4);
    size_t o_hb    = take((size_t)2 * 256 * 256 * 64 * 2);
    size_t o_hb1   = take((size_t)2 * 128 * 128 * 64 * 2);
    size_t o_hb2   = take((size_t)2 * 64 * 64 * 64 * 2);
    size_t o_c1    = take((size_t)2 * 128 * 128 * 64 * 4);
    size_t o_c2    = take((size_t)2 * 64 * 64 * 64 * 4);
    size_t o_m1b   = take((size_t)32768 * 64 * 2);
    size_t o_m2b   = take((size_t)32768 * 128 * 2);
    size_t o_phi   = take((size_t)32768 * 256 * 4);     // + o_psi region reused as qh1
    size_t o_psi   = take((size_t)32768 * 256 * 4);
    size_t o_x2    = take((size_t)2 * 128 * 128 * 64 * 4);
    size_t o_y     = take((size_t)2 * 64 * 4 * 4);

    unsigned short* pk_conv = (unsigned short*)(ws + o_convW);
    unsigned short* pk_phi2 = (unsigned short*)(ws + o_phi2W);
    unsigned short* pk_phi3 = (unsigned short*)(ws + o_phi3W);
    unsigned short* pk_psi2 = (unsigned short*)(ws + o_psi2W);
    unsigned short* pk_psi3 = (unsigned short*)(ws + o_psi3W);
    unsigned short* pk_q1   = (unsigned short*)(ws + o_q1W);
    float* hbuf  = (float*)(ws + o_h);
    float* x1buf = (float*)(ws + o_x1);
    unsigned short* hb  = (unsigned short*)(ws + o_hb);
    unsigned short* hb1 = (unsigned short*)(ws + o_hb1);
    unsigned short* hb2 = (unsigned short*)(ws + o_hb2);
    float* c1 = (float*)(ws + o_c1);
    float* c2 = (float*)(ws + o_c2);
    unsigned short* m1b = (unsigned short*)(ws + o_m1b);
    unsigned short* m2b = (unsigned short*)(ws + o_m2b);
    float* phibuf = (float*)(ws + o_phi);
    float* psibuf = (float*)(ws + o_psi);
    float* qh1    = (float*)(ws + o_phi);              // reuse phi+psi (67 MB)
    float* x2lo   = (float*)(ws + o_x2);
    float* ybuf   = (float*)(ws + o_y);

    // ---- weight prepack (bf16) ----
    k_pack_conv<<<(4 * 3 * 64 * 64 * 9 + 255) / 256, 256, 0, stream>>>(convW, pk_conv);
    for (int i = 0; i < 4; ++i) {
        k_pack_mat<<<(64 * 128 + 255) / 256, 256, 0, stream>>>(
            phiW2 + (size_t)i * 64 * 128, pk_phi2 + (size_t)i * 128 * 64, 64, 128);
        k_pack_mat<<<(128 * 256 + 255) / 256, 256, 0, stream>>>(
            phiW3 + (size_t)i * 128 * 256, pk_phi3 + (size_t)i * 256 * 128, 128, 256);
        k_pack_mat<<<(64 * 128 + 255) / 256, 256, 0, stream>>>(
            psiW2 + (size_t)i * 64 * 128, pk_psi2 + (size_t)i * 128 * 64, 64, 128);
        k_pack_mat<<<(128 * 256 + 255) / 256, 256, 0, stream>>>(
            psiW3 + (size_t)i * 128 * 256, pk_psi3 + (size_t)i * 256 * 128, 128, 256);
    }
    k_pack_mat<<<(64 * 128 + 255) / 256, 256, 0, stream>>>(q1W, pk_q1, 64, 128);

    const size_t NPTS = (size_t)2 * 256 * 256;          // 131072
    const size_t NPTS_D = (size_t)2 * 128 * 128;        // 32768

    // ---- lift: h = xa @ pW + pb ----
    k_lift<<<dim3((unsigned)NPTS), 64, 0, stream>>>(a, x, pW, pb, hbuf);

    // ---- 4 operator blocks ----
    for (int i = 0; i < 4; ++i) {
        // bf16 copies of h at 3 resolutions
        k_cvt_ds<<<dim3((unsigned)NPTS), 64, 0, stream>>>(hbuf, hb, 256, 1);
        k_cvt_ds<<<dim3((unsigned)NPTS_D), 64, 0, stream>>>(hbuf, hb1, 128, 2);
        k_cvt_ds<<<dim3((unsigned)(2 * 64 * 64)), 64, 0, stream>>>(hbuf, hb2, 64, 4);

        // multilevel conv (WMMA)
        const unsigned short* cw0 = pk_conv + (size_t)(i * 3 + 0) * 64 * 9 * 64;
        const unsigned short* cw1 = pk_conv + (size_t)(i * 3 + 1) * 64 * 9 * 64;
        const unsigned short* cw2 = pk_conv + (size_t)(i * 3 + 2) * 64 * 9 * 64;
        k_conv3x3_wmma<<<dim3(16, 256, 2), 128, 0, stream>>>(hb,  cw0, convb + (i * 3 + 0) * 64, x1buf, 256);
        k_conv3x3_wmma<<<dim3(8, 128, 2),  128, 0, stream>>>(hb1, cw1, convb + (i * 3 + 1) * 64, c1,    128);
        k_conv3x3_wmma<<<dim3(4, 64, 2),   128, 0, stream>>>(hb2, cw2, convb + (i * 3 + 2) * 64, c2,    64);
        k_upsample_add<<<dim3((unsigned)NPTS), 64, 0, stream>>>(c1, x1buf, 128);
        k_upsample_add<<<dim3((unsigned)NPTS), 64, 0, stream>>>(c2, x1buf, 64);

        // low-rank branch: phi
        k_mlp1<<<dim3((unsigned)NPTS_D), 64, 0, stream>>>(a, x, phiW1 + (size_t)i * 3 * 64,
                                                          phib1 + (size_t)i * 64, m1b);
        k_gemm_wmma<<<dim3(2048, 2), 128, 0, stream>>>(m1b, pk_phi2 + (size_t)i * 128 * 64,
                                                       phib2 + (size_t)i * 128, nullptr, m2b, 128, 64, 1);
        k_gemm_wmma<<<dim3(2048, 4), 128, 0, stream>>>(m2b, pk_phi3 + (size_t)i * 256 * 128,
                                                       phib3 + (size_t)i * 256, phibuf, nullptr, 256, 128, 0);
        // psi
        k_mlp1<<<dim3((unsigned)NPTS_D), 64, 0, stream>>>(a, x, psiW1 + (size_t)i * 3 * 64,
                                                          psib1 + (size_t)i * 64, m1b);
        k_gemm_wmma<<<dim3(2048, 2), 128, 0, stream>>>(m1b, pk_psi2 + (size_t)i * 128 * 64,
                                                       psib2 + (size_t)i * 128, nullptr, m2b, 128, 64, 1);
        k_gemm_wmma<<<dim3(2048, 4), 128, 0, stream>>>(m2b, pk_psi3 + (size_t)i * 256 * 128,
                                                       psib3 + (size_t)i * 256, psibuf, nullptr, 256, 128, 0);

        // rank-4 contractions
        k_reduce_y<<<dim3(64, 2), 256, 0, stream>>>(psibuf, hbuf, ybuf);
        k_x2<<<dim3((unsigned)NPTS_D), 64, 0, stream>>>(phibuf, ybuf, x2lo);

        // combine + layernorm (+relu except last block)
        k_combine_ln<<<dim3((unsigned)NPTS), 64, 0, stream>>>(x1buf, x2lo,
                                                              lnG + (size_t)i * 64, lnB + (size_t)i * 64,
                                                              hbuf, (i != 3) ? 1 : 0);
    }

    // ---- head: q1 (WMMA) then q2 dot ----
    k_cvt_ds<<<dim3((unsigned)NPTS), 64, 0, stream>>>(hbuf, hb, 256, 1);
    k_gemm_wmma<<<dim3(8192, 2), 128, 0, stream>>>(hb, pk_q1, q1b, qh1, nullptr, 128, 64, 1);
    k_q2<<<dim3((unsigned)((NPTS + 255) / 256)), 256, 0, stream>>>(qh1, q2W, q2b, out, (int)NPTS);
}
